// XMem_2310692405398
// MI455X (gfx1250) — matrix operand order repaired and checked by reading, not compiled, not run on hardware
//
#include <hip/hip_runtime.h>

// ---------------------------------------------------------------------------
// XMem memory-attention readout, fused flash-style, bf16 WMMA (gfx1250)
//   pass 1: per-column online max/sum-exp (waves on disjoint N-chunks)
//   pass 2: P staged via LDS in WMMA-B layout; waves own 128 value rows each
// ---------------------------------------------------------------------------

typedef __bf16 bf16_t;
typedef __attribute__((ext_vector_type(16))) __bf16 v16bf;
typedef __attribute__((ext_vector_type(2)))  __bf16 v2bf;
typedef __attribute__((ext_vector_type(8)))  float  v8f;

constexpr int BB  = 2;
constexpr int CK  = 64;
constexpr int CV  = 512;
constexpr int OO  = 2;
constexpr int TT  = 16;
constexpr int HH  = 30;
constexpr int WW  = 54;
constexpr int HW  = HH * WW;            // 1620 queries
constexpr int NN  = TT * HW;            // 25920 memory slots
constexpr int VD  = OO * CV;            // 1024 value channels
constexpr int QTILES = (HW + 15) / 16;  // 102
constexpr int NSUPER = (NN + 255) / 256;
constexpr float SCALE = 0.125f;         // 1/sqrt(CK)
constexpr float NEG_INF = -3.0e38f;

static __device__ __forceinline__ bf16_t f2bf(float x) { return (bf16_t)x; }

static __device__ __forceinline__ unsigned pack2bf(float a, float b) {
    union { v2bf v; unsigned u; } u;
    u.v[0] = (bf16_t)a;
    u.v[1] = (bf16_t)b;
    return u.u;
}

__global__ __launch_bounds__(256) void xmem_readout_kernel(
    const float* __restrict__ qk,   // [B, CK, HW]
    const float* __restrict__ qe,   // [B, CK, HW]
    const float* __restrict__ mk,   // [B, CK, N]
    const float* __restrict__ ms,   // [B, N]
    const float* __restrict__ mv,   // [B, VD, N]
    float* __restrict__ out)        // [B, VD, HW]
{
    const int tid  = threadIdx.x;
    const int lane = tid & 31;
    const int wave = tid >> 5;
    const int h    = lane >> 4;       // lane half (WMMA layout)
    const int lq   = lane & 15;       // column within 16-wide tile
    const int b    = blockIdx.y;
    const int q0   = blockIdx.x * 16;
    const int qcol = min(q0 + lq, HW - 1);   // clamp tail columns

    const float* qkb = qk + (size_t)b * CK * HW;
    const float* qeb = qe + (size_t)b * CK * HW;
    const float* mkb = mk + (size_t)b * CK * NN;
    const float* msb = ms + (size_t)b * NN;
    const float* mvb = mv + (size_t)b * VD * NN;
    float*       oub = out + (size_t)b * VD * HW;

    // ---------------- prologue: sim B-fragments (K=128 folded) + b_sq ------
    // B-matrix, k order chosen here (not ISA table order): lane half h holds K = 16h + {2j,2j+1}
    // in packed VGPR j.  Folded K space: k=2c -> 2*qk*qe ; k=2c+1 -> -qe.
    v16bf Bs[4];
    float bsq = 0.f;
#pragma unroll
    for (int kf = 0; kf < 4; ++kf) {
#pragma unroll
        for (int j = 0; j < 8; ++j) {
            const int c = kf * 16 + 8 * h + j;
            const float qkv = qkb[c * HW + qcol];
            const float qev = qeb[c * HW + qcol];
            Bs[kf][2 * j]     = f2bf(2.f * qkv * qev);
            Bs[kf][2 * j + 1] = f2bf(-qev);
            bsq += qev * qkv * qkv;
        }
    }
    bsq += __shfl_xor(bsq, 16);     // each lane half saw 32 of the 64 channels

    // sim-core tile for 16 memory rows starting at n0:
    // A-matrix, same k order as Bs (not ISA table order): lane holds row n0+lq; VGPR j packs
    // K = 16h + 2j  (and +1) -> channel c = K/2.
    auto sim_tile = [&](int n0) -> v8f {
        v8f c = {};
#pragma unroll
        for (int kf = 0; kf < 4; ++kf) {
            v16bf A;
#pragma unroll
            for (int j = 0; j < 8; ++j) {
                const int cch = kf * 16 + 8 * h + j;
                const float mkv = mkb[(size_t)cch * NN + n0 + lq];
                A[2 * j]     = f2bf(mkv);
                A[2 * j + 1] = f2bf(mkv * mkv);
            }
            c = __builtin_amdgcn_wmma_f32_16x16x32_bf16(
                    false, A, false, Bs[kf], (short)0, c, false, false);
        }
        return c;
    };

    // ---------------- pass 1: per-column running max & sum-exp -------------
    float m_run = NEG_INF, s_run = 0.f;
    for (int ch = wave; ch < NN / 16; ch += 8) {
        const int n0 = ch * 16;
        const v8f cs = sim_tile(n0);
        const float msv = msb[n0 + lq];           // lane r holds ms row r
        float sv[8];
        float mloc = NEG_INF;
#pragma unroll
        for (int i = 0; i < 8; ++i) {             // C VGPR i -> row i + 8h
            const float msr = __shfl(msv, i + 8 * h);
            sv[i] = (cs[i] - bsq) * msr * SCALE;
            mloc = fmaxf(mloc, sv[i]);
        }
        const float mnew = fmaxf(m_run, mloc);
        float add = 0.f;
#pragma unroll
        for (int i = 0; i < 8; ++i) add += __expf(sv[i] - mnew);
        s_run = s_run * __expf(m_run - mnew) + add;
        m_run = mnew;
    }
    {   // merge the two lane halves (same q column)
        const float m2 = __shfl_xor(m_run, 16);
        const float s2 = __shfl_xor(s_run, 16);
        const float mm = fmaxf(m_run, m2);
        s_run = s_run * __expf(m_run - mm) + s2 * __expf(m2 - mm);
        m_run = mm;
    }

    __shared__ float red_m[8][16];
    __shared__ float red_s[8][16];
    __shared__ float Mq[16];
    __shared__ float iSq[16];
    // P staging: [sub-chunk][pair-row][q], stride 20 dwords => conflict-free
    __shared__ unsigned Pl[8][16][20];

    if (lane < 16) { red_m[wave][lq] = m_run; red_s[wave][lq] = s_run; }
    __syncthreads();
    if (tid < 16) {
        float M = NEG_INF;
        for (int w = 0; w < 8; ++w) M = fmaxf(M, red_m[w][tid]);
        float S = 0.f;
        for (int w = 0; w < 8; ++w) S += red_s[w][tid] * __expf(red_m[w][tid] - M);
        Mq[tid]  = M;
        iSq[tid] = 1.f / S;
    }
    __syncthreads();
    const float Mfin = Mq[lq];
    const float invS = iSq[lq];

    // ---------------- pass 2: readout out[v,q] += V[v,n] * P[n,q] ----------
    v8f acc[8];
#pragma unroll
    for (int vt = 0; vt < 8; ++vt) acc[vt] = (v8f){};
    const int v0w = wave * 128;                   // this wave's value rows

    for (int sc = 0; sc < NSUPER; ++sc) {
        const int nbase = sc * 256;
        const int nsub  = min(8, (NN - nbase) / 32);

        // ---- produce: wave w computes softmaxed P for sub-chunk w ----
        if (wave < nsub) {
            const int n0 = nbase + wave * 32;
            const v8f c0 = sim_tile(n0);
            const v8f c1 = sim_tile(n0 + 16);
            const float msv = msb[n0 + lane];     // lane r holds ms row r (0..31)
            float p0[8], p1[8];
#pragma unroll
            for (int i = 0; i < 8; ++i) {
                const int r = i + 8 * h;
                const float ms0 = __shfl(msv, r);
                const float ms1 = __shfl(msv, 16 + r);
                p0[i] = __expf((c0[i] - bsq) * ms0 * SCALE - Mfin);
                p1[i] = __expf((c1[i] - bsq) * ms1 * SCALE - Mfin);
            }
            // lane rows are pair-aligned: tile0 rows 8h+2m,+1 -> pair-row 4h+m
            //                             tile1 rows 16+8h+2m -> pair-row 8+4h+m
#pragma unroll
            for (int m = 0; m < 4; ++m) {
                Pl[wave][4 * h + m][lq]     = pack2bf(p0[2 * m], p0[2 * m + 1]);
                Pl[wave][8 + 4 * h + m][lq] = pack2bf(p1[2 * m], p1[2 * m + 1]);
            }
        }
        __syncthreads();

        // ---- consume: every wave reads P fragments, streams its V rows ----
        for (int s8 = 0; s8 < nsub; ++s8) {
            const int n0 = nbase + s8 * 32;
            // B-fragment: VGPR j = pair-row (j&3) + 4h + 8*(j>>2) (matches Va k order)
            union { v16bf v; unsigned u[8]; } P;
#pragma unroll
            for (int j = 0; j < 8; ++j) P.u[j] = Pl[s8][(j & 3) + 4 * h + 8 * (j >> 2)][lq];

#pragma unroll
            for (int vt = 0; vt < 8; ++vt) {
                const float* vp =
                    mvb + (size_t)(v0w + vt * 16 + lq) * NN + n0 + 8 * h;
                const float4 r0  = *(const float4*)(vp);
                const float4 r0b = *(const float4*)(vp + 4);
                const float4 r1  = *(const float4*)(vp + 16);
                const float4 r1b = *(const float4*)(vp + 20);
                v16bf Va;
                Va[0]  = f2bf(r0.x);  Va[1]  = f2bf(r0.y);
                Va[2]  = f2bf(r0.z);  Va[3]  = f2bf(r0.w);
                Va[4]  = f2bf(r0b.x); Va[5]  = f2bf(r0b.y);
                Va[6]  = f2bf(r0b.z); Va[7]  = f2bf(r0b.w);
                Va[8]  = f2bf(r1.x);  Va[9]  = f2bf(r1.y);
                Va[10] = f2bf(r1.z);  Va[11] = f2bf(r1.w);
                Va[12] = f2bf(r1b.x); Va[13] = f2bf(r1b.y);
                Va[14] = f2bf(r1b.z); Va[15] = f2bf(r1b.w);
                acc[vt] = __builtin_amdgcn_wmma_f32_16x16x32_bf16(
                              false, Va, false, P.v, (short)0, acc[vt],
                              false, false);
            }
        }
        __syncthreads();   // protect Pl before next super-chunk overwrites
    }

    // ---------------- epilogue ---------------------------------------------
    if (q0 + lq < HW) {
#pragma unroll
        for (int vt = 0; vt < 8; ++vt) {
#pragma unroll
            for (int i = 0; i < 8; ++i) {
                const int vrow = v0w + vt * 16 + i + 8 * h;
                oub[(size_t)vrow * HW + q0 + lq] = acc[vt][i] * invS;
            }
        }
    }
}

extern "C" void kernel_launch(void* const* d_in, const int* in_sizes, int n_in,
                              void* d_out, int out_size, void* d_ws, size_t ws_size,
                              hipStream_t stream) {
    const float* qk = (const float*)d_in[0];   // query_key      [B,CK,H,W]
    const float* qe = (const float*)d_in[1];   // query_selection[B,CK,H,W]
    const float* mk = (const float*)d_in[2];   // memory_key     [B,CK,T,H,W]
    const float* ms = (const float*)d_in[3];   // memory_shrink  [B,1,T,H,W]
    const float* mv = (const float*)d_in[4];   // memory_value   [B,O,CV,T,H,W]
    float* out = (float*)d_out;                // [B,O,CV,H,W]

    dim3 grid(QTILES, BB);
    xmem_readout_kernel<<<grid, 256, 0, stream>>>(qk, qe, mk, ms, mv, out);
}
